// NativeBitLinear_19799799235326
// MI455X (gfx1250) — compile-verified, matrix operand order checked by reading, not run on hardware
//
#include <hip/hip_runtime.h>
#include <hip/hip_bf16.h>
#include <stdint.h>

// ---------------- problem sizes (fixed by reference) ----------------
#define IN_F    4096              // K
#define OUT_F   4096              // N
#define M_ROWS  4096              // 2 * 2048
#define NENT    8

// ---------------- GEMM tiling ----------------
#define BM 128
#define BN 128
#define BK 32
#define LDS_K 48                  // padded K-stride: 64B data + 32B pad = 96B rows

typedef __bf16 bf16_t;
typedef __attribute__((ext_vector_type(16))) __bf16 v16bf;
typedef __attribute__((ext_vector_type(8)))  __bf16 v8bf;
typedef __attribute__((ext_vector_type(4)))  __bf16 v4bf;
typedef __attribute__((ext_vector_type(8)))  float  v8f;
typedef __attribute__((ext_vector_type(4)))  float  v4f;
typedef __attribute__((ext_vector_type(4)))  unsigned int v4u;
typedef __attribute__((ext_vector_type(8)))  int    v8i;
typedef __attribute__((ext_vector_type(4)))  int    v4i;

#if defined(__has_builtin)
#if __has_builtin(__builtin_amdgcn_tensor_load_to_lds)
#define HAVE_TDM 1
#endif
#endif

// ---- fallback: per-lane async global->LDS copy (ASYNCcnt), 16B/lane ----
__device__ __forceinline__ void async_copy_b128(uint32_t lds_byte_addr,
                                                const void* gaddr) {
  asm volatile("global_load_async_to_lds_b128 %0, %1, off"
               :: "v"(lds_byte_addr), "v"((uint64_t)(uintptr_t)gaddr)
               : "memory");
}
__device__ __forceinline__ void wait_async0() {
  asm volatile("s_wait_asynccnt 0" ::: "memory");
}

#ifdef HAVE_TDM
// ---- CDNA5 Tensor Data Mover: one descriptor moves a whole 2D tile ----
// tile = tile_rows x BK bf16 elements, source rows strided by IN_F elems,
// dest rows padded to LDS_K elems via D# pad fields (64B data + 32B pad).
__device__ __forceinline__ void tdm_load_tile(const void* gaddr,
                                              uint32_t lds_addr,
                                              int tile_rows) {
  uint64_t ga = (uint64_t)(uintptr_t)gaddr;
  v4u g0;
  g0[0] = 1u;                                        // count=1 (valid), user
  g0[1] = lds_addr;                                  // LDS byte address
  g0[2] = (uint32_t)ga;                              // global addr [31:0]
  g0[3] = (uint32_t)((ga >> 32) & 0x01FFFFFFu)       // global addr [56:32]
          | (2u << 30);                              // type = 2 ("image")
  v8i g1;
  g1[0] = (1 << 16)                                  // data_size = 2 bytes
        | (1 << 20)                                  // pad_enable
        | (3 << 22)                                  // pad_interval: 16 DW (64B)
        | (7 << 25);                                 // pad_amount: 8 DW (32B)
  g1[1] = (int)((IN_F & 0xFFFF) << 16);              // tensor_dim0 lo16 @48
  g1[2] = (int)(((IN_F >> 16) & 0xFFFF)              // tensor_dim0 hi16
        | ((M_ROWS & 0xFFFF) << 16));                // tensor_dim1 lo16
  g1[3] = (int)(((M_ROWS >> 16) & 0xFFFF)            // tensor_dim1 hi16
        | ((BK & 0xFFFF) << 16));                    // tile_dim0 = 32 elems
  g1[4] = tile_rows & 0xFFFF;                        // tile_dim1 (tile_dim2=0)
  g1[5] = IN_F;                                      // tensor_dim0_stride lo32
  g1[6] = 0;
  g1[7] = 0;
  v4i z4 = {0, 0, 0, 0};
#if __clang_major__ >= 23
  v8i z8 = {0, 0, 0, 0, 0, 0, 0, 0};
  __builtin_amdgcn_tensor_load_to_lds(g0, g1, z4, z4, z8, 0);
#else
  __builtin_amdgcn_tensor_load_to_lds(g0, g1, z4, z4, 0);
#endif
}
#endif

// =====================================================================
// Kernel 1: per-block VQ quantization of W -> bf16 (same [N,K] layout)
// =====================================================================
__global__ __launch_bounds__(256)
void vq_quant_kernel(const float* __restrict__ w,
                     const float* __restrict__ cb,   // [NUM_BLOCKS, 8]
                     bf16_t* __restrict__ qw) {
  int e  = blockIdx.x * blockDim.x + threadIdx.x;    // one weight element
  int nb = e >> 6;                                   // block id (64/block)
  float wv = w[e];
  v4f c0 = *(const v4f*)(cb + (size_t)nb * NENT);
  v4f c1 = *(const v4f*)(cb + (size_t)nb * NENT + 4);
  float cr[8] = {c0[0], c0[1], c0[2], c0[3], c1[0], c1[1], c1[2], c1[3]};
  float bd = (wv - cr[0]) * (wv - cr[0]);
  float bv = cr[0];
#pragma unroll
  for (int t = 1; t < 8; ++t) {
    float d = (wv - cr[t]) * (wv - cr[t]);
    if (d < bd) { bd = d; bv = cr[t]; }              // first-min wins (ties)
  }
  qw[e] = (bf16_t)bv;
}

// =====================================================================
// Kernel 2: x fp32 -> bf16 (4 elems/thread, vectorized)
// =====================================================================
__global__ __launch_bounds__(256)
void x_to_bf16_kernel(const float* __restrict__ x, bf16_t* __restrict__ xb) {
  size_t i = ((size_t)blockIdx.x * blockDim.x + threadIdx.x) * 4;
  v4f v = *(const v4f*)(x + i);
  v4bf o;
  o[0] = (bf16_t)v[0]; o[1] = (bf16_t)v[1];
  o[2] = (bf16_t)v[2]; o[3] = (bf16_t)v[3];
  *(v4bf*)(xb + i) = o;
}

// =====================================================================
// Kernel 3: bf16 WMMA GEMM  out[M,N] = Xb[M,K] * Qw[N,K]^T + bias[N]
//   256 threads = 8 waves, tile 128x128, BK=32, double-buffered LDS,
//   staging via TDM tensor_load_to_lds (TENSORcnt) — one descriptor per
//   tile issued by wave 0 (A) / wave 1 (B) — async-to-LDS fallback.
//   per-wave 64x32 => 4x2 v_wmma_f32_16x16x32_bf16 accumulators.
// =====================================================================
__global__ __launch_bounds__(256)
void gemm_bf16_wmma_kernel(const bf16_t* __restrict__ A,   // [M, K]
                           const bf16_t* __restrict__ B,   // [N, K]
                           const float*  __restrict__ bias,
                           float* __restrict__ out) {
  __shared__ bf16_t sA[2][BM * LDS_K];
  __shared__ bf16_t sB[2][BN * LDS_K];

  const int tid   = threadIdx.x;
  const int lane  = tid & 31;
  const int wave  = tid >> 5;
  const int waveM = wave >> 2;            // 0..1  (64 rows each)
  const int waveN = wave & 3;             // 0..3  (32 cols each)
  const int m0 = blockIdx.y * BM;
  const int n0 = blockIdx.x * BN;

  v8f acc[4][2] = {};                     // 4 M-frags x 2 N-frags

  // stage one K-slab (BK cols) of A and B tiles into LDS buffer `buf`
  auto stage = [&](int buf, int k0) {
#ifdef HAVE_TDM
    if (wave == 0)
      tdm_load_tile(A + (size_t)m0 * IN_F + k0,
                    (uint32_t)(uintptr_t)&sA[buf][0], BM);
    else if (wave == 1)
      tdm_load_tile(B + (size_t)n0 * IN_F + k0,
                    (uint32_t)(uintptr_t)&sB[buf][0], BN);
#else
#pragma unroll
    for (int c = 0; c < 2; ++c) {
      int chunk = tid + c * 256;
      int row   = chunk >> 2;             // 4 chunks per row
      int col   = (chunk & 3) * 8;
      uint32_t la = (uint32_t)(uintptr_t)&sA[buf][row * LDS_K + col];
      async_copy_b128(la, A + (size_t)(m0 + row) * IN_F + k0 + col);
      uint32_t lb = (uint32_t)(uintptr_t)&sB[buf][row * LDS_K + col];
      async_copy_b128(lb, B + (size_t)(n0 + row) * IN_F + k0 + col);
    }
#endif
  };

  auto drain_stage = [&] {
#ifdef HAVE_TDM
    if (wave < 2) __builtin_amdgcn_s_wait_tensorcnt(0);
#else
    wait_async0();
#endif
    __syncthreads();
  };

  // load a 16x32 bf16 fragment (A-layout; B identical since B is K-major):
  // lane L: row = L%16; half=L/16 -> K octets {half*8, 16+half*8}
  auto load_frag = [&](const bf16_t* base, int row) -> v16bf {
    const bf16_t* p = base + (size_t)(row + (lane & 15)) * LDS_K
                           + (lane >> 4) * 8;
    v8bf lo = *(const v8bf*)p;            // ds_load_b128
    v8bf hi = *(const v8bf*)(p + 16);     // ds_load_b128
    return __builtin_shufflevector(lo, hi, 0, 1, 2, 3, 4, 5, 6, 7,
                                           8, 9, 10, 11, 12, 13, 14, 15);
  };

  const int NSTAGE = IN_F / BK;           // 128

  stage(0, 0);
  drain_stage();

  for (int kt = 0; kt < NSTAGE; ++kt) {
    const int buf = kt & 1;
    if (kt + 1 < NSTAGE) stage(buf ^ 1, (kt + 1) * BK);
    if (kt + 2 < NSTAGE)                  // warm L2 for the slab after next
      __builtin_prefetch(A + (size_t)(m0 + (tid >> 1)) * IN_F
                           + (kt + 2) * BK, 0, 1);

    v16bf af[4], bfg[2];
#pragma unroll
    for (int i = 0; i < 4; ++i)
      af[i] = load_frag(&sA[buf][0], waveM * 64 + i * 16);
#pragma unroll
    for (int j = 0; j < 2; ++j)
      bfg[j] = load_frag(&sB[buf][0], waveN * 32 + j * 16);

#pragma unroll
    for (int i = 0; i < 4; ++i)
#pragma unroll
      for (int j = 0; j < 2; ++j)
        acc[i][j] = __builtin_amdgcn_wmma_f32_16x16x32_bf16(
            false, af[i], false, bfg[j], (short)0, acc[i][j], false, false);

    drain_stage();
  }

  // epilogue: C/D layout -> VGPR v holds M = 16*i + 8*(lane>>4) + v, N = lane&15
#pragma unroll
  for (int i = 0; i < 4; ++i) {
#pragma unroll
    for (int j = 0; j < 2; ++j) {
      int n  = n0 + waveN * 32 + j * 16 + (lane & 15);
      float bv = bias[n];
      int mb = m0 + waveM * 64 + i * 16 + (lane >> 4) * 8;
#pragma unroll
      for (int v = 0; v < 8; ++v)
        out[(size_t)(mb + v) * OUT_F + n] = acc[i][j][v] + bv;
    }
  }
}

// =====================================================================
extern "C" void kernel_launch(void* const* d_in, const int* in_sizes, int n_in,
                              void* d_out, int out_size, void* d_ws, size_t ws_size,
                              hipStream_t stream) {
  const float* x    = (const float*)d_in[0];   // [2,2048,4096] fp32
  const float* w    = (const float*)d_in[1];   // [4096,4096]   fp32
  const float* bias = (const float*)d_in[2];   // [4096]        fp32
  const float* cb   = (const float*)d_in[3];   // [262144,8]    fp32
  float* out = (float*)d_out;                  // [2,2048,4096] fp32

  bf16_t* qw = (bf16_t*)d_ws;                                        // 32 MB
  bf16_t* xb = (bf16_t*)((char*)d_ws + (size_t)OUT_F * IN_F * 2);    // 32 MB

  (void)in_sizes; (void)n_in; (void)out_size; (void)ws_size;

  // 1) VQ-quantize weights -> bf16 [N,K]
  vq_quant_kernel<<<(OUT_F * IN_F) / 256, 256, 0, stream>>>(w, cb, qw);
  // 2) cast activations -> bf16 [M,K]
  x_to_bf16_kernel<<<((size_t)M_ROWS * IN_F) / (256 * 4), 256, 0, stream>>>(x, xb);
  // 3) WMMA GEMM + bias
  dim3 grid(OUT_F / BN, M_ROWS / BM);
  gemm_bf16_wmma_kernel<<<grid, 256, 0, stream>>>(xb, qw, bias, out);
}
